// _QuantizedLinear_20160576487470
// MI455X (gfx1250) — matrix operand order repaired and checked by reading, not compiled, not run on hardware
//
#include <hip/hip_runtime.h>
#include <stdint.h>

// ---------------------------------------------------------------------------
// Quantized linear as two passes:
//   pass 1: X fp32 -> f16, W int32 -> f16 (exact for int8 range) into d_ws
//   pass 2: f16 WMMA GEMM, tiles streamed into LDS by the Tensor Data Mover
//           (tensor_load_to_lds), double-buffered, dequant epilogue in fp32.
// M = B*S = 8192, N = D_OUT = 4096, K = D_IN = 4096.
// ---------------------------------------------------------------------------

typedef __attribute__((ext_vector_type(16))) _Float16 v16h;
typedef __attribute__((ext_vector_type(8)))  _Float16 v8h;
typedef __attribute__((ext_vector_type(4)))  _Float16 v4h;
typedef __attribute__((ext_vector_type(8)))  float    v8f;
typedef __attribute__((ext_vector_type(4)))  float    v4f;
typedef __attribute__((ext_vector_type(4)))  int      v4i;
typedef __attribute__((ext_vector_type(4)))  unsigned int u32x4;
typedef __attribute__((ext_vector_type(8)))  int      i32x8;
typedef __attribute__((ext_vector_type(4)))  int      i32x4;

#define D_IN    4096
#define D_OUT   4096
#define M_TOTAL 8192

// ---------------- TDM GEMM tiling ----------------
#define TBM 128            // block tile M
#define TBN 128            // block tile N
#define TBK 32             // K tile == one WMMA k-step
#define TLDSK (TBK + 8)    // TDM pads each 32-half row by 8 halves (16B)

// ---------------- fallback (fused) tiling --------
#define FBM 128
#define FBN 128
#define FBK 64
#define FLDSK (FBK + 8)

// Get the 32-bit LDS byte address of a __shared__ object (generic -> AS(3)).
__device__ __forceinline__ uint32_t lds_addr_of(const void* p) {
    return (uint32_t)(uintptr_t)(__attribute__((address_space(3))) const void*)p;
}

// Issue one 2D TDM tile load: tile_rows x TBK f16 elements from a row-major
// [tensor_rows x D_IN] f16 tensor into LDS with 16B padding per row.
// D# layout per CDNA5 ISA 8.3/8.4. 6-arg builtin (clang-23 signature).
__device__ __forceinline__ void tdm_load_tile_f16(const _Float16* gptr,
                                                  uint32_t lds_byte_addr,
                                                  uint32_t tensor_rows,
                                                  uint32_t tile_rows)
{
    const uint64_t ga = (uint64_t)(uintptr_t)gptr;

    u32x4 g0;
    g0.x = 1u;                                           // count=1, user desc
    g0.y = lds_byte_addr;                                // lds_addr
    g0.z = (uint32_t)ga;                                 // global_addr[31:0]
    g0.w = ((uint32_t)(ga >> 32) & 0x01FFFFFFu)          // global_addr[56:32]
         | (2u << 30);                                   // type = 2 ("image")

    i32x8 g1;
    g1[0] = (int)((1u << 16)      // data_size = 2 bytes
                | (1u << 20)      // pad_enable
                | (3u << 22)      // pad_interval: code 3 = 16 DWORDs (= 64B row)
                | (3u << 25));    // pad_amount:   code 3 = 4 DWORDs  (= 16B)
    g1[1] = (int)(((uint32_t)D_IN & 0xFFFFu) << 16);     // tensor_dim0[15:0]
    g1[2] = (int)((((uint32_t)D_IN >> 16) & 0xFFFFu)     // tensor_dim0[31:16]
                | ((tensor_rows & 0xFFFFu) << 16));      // tensor_dim1[15:0]
    g1[3] = (int)(((tensor_rows >> 16) & 0xFFFFu)        // tensor_dim1[31:16]
                | ((uint32_t)TBK << 16));                // tile_dim0 = 32
    g1[4] = (int)(tile_rows & 0xFFFFu);                  // tile_dim1; tile_dim2=0
    g1[5] = (int)(uint32_t)D_IN;                         // tensor_dim0_stride[31:0]
    g1[6] = 0;                                           // stride hi / dim1_stride lo
    g1[7] = 0;

    const i32x4 gz4 = {0, 0, 0, 0};                      // groups 2/3: unused (2D)
    const i32x8 gz8 = {0, 0, 0, 0, 0, 0, 0, 0};          // trailing group: unused
    __builtin_amdgcn_tensor_load_to_lds(g0, g1, gz4, gz4, gz8, 0);
}

// ---------------------------------------------------------------------------
// Pass 1: elementwise conversions into workspace.
// ---------------------------------------------------------------------------
__global__ __launch_bounds__(256)
void cvt_f32_to_f16(const float* __restrict__ src, _Float16* __restrict__ dst, int n4)
{
    const int i = blockIdx.x * 256 + threadIdx.x;
    if (i < n4) {
        const v4f v = *(const v4f*)(src + (size_t)i * 4);
        v4h h;
        h.x = (_Float16)v.x; h.y = (_Float16)v.y;
        h.z = (_Float16)v.z; h.w = (_Float16)v.w;
        *(v4h*)(dst + (size_t)i * 4) = h;
    }
}

__global__ __launch_bounds__(256)
void cvt_i32_to_f16(const int* __restrict__ src, _Float16* __restrict__ dst, int n4)
{
    const int i = blockIdx.x * 256 + threadIdx.x;
    if (i < n4) {
        const v4i v = *(const v4i*)(src + (size_t)i * 4);
        v4h h;
        h.x = (_Float16)v.x; h.y = (_Float16)v.y;
        h.z = (_Float16)v.z; h.w = (_Float16)v.w;
        *(v4h*)(dst + (size_t)i * 4) = h;
    }
}

// ---------------------------------------------------------------------------
// Pass 2: TDM-fed, double-buffered f16 WMMA GEMM.
// 256 threads = 8 waves; wave (wm,wn) owns a 64x32 subtile = 4x2 WMMA accums.
// ---------------------------------------------------------------------------
__global__ __launch_bounds__(256)
void qlinear_wmma_tdm(const _Float16* __restrict__ xh,
                      const _Float16* __restrict__ wh,
                      const float* __restrict__ scale,
                      const float* __restrict__ bias,
                      float*       __restrict__ out)
{
    __shared__ _Float16 sX[2][TBM][TLDSK];   // 2 x 128 x 40 halves = 20 KB
    __shared__ _Float16 sW[2][TBN][TLDSK];   // 2 x 128 x 40 halves = 20 KB

    const int tid   = threadIdx.x;
    const int lane  = tid & 31;
    const int wave  = tid >> 5;
    const int wm    = wave >> 2;     // 0..1
    const int wn    = wave & 3;      // 0..3
    const int lhalf = lane >> 4;     // 0/1
    const int l16   = lane & 15;

    const int m0 = blockIdx.y * TBM;
    const int n0 = blockIdx.x * TBN;

    float sc[2], bi[2];
#pragma unroll
    for (int j = 0; j < 2; ++j) {
        const int n = n0 + wn * 32 + j * 16 + l16;
        sc[j] = scale[n];
        bi[j] = bias[n];
    }

    v8f acc[4][2] = {};

    const int NIT = D_IN / TBK;   // 128

    // Prologue: queue tiles 0 and 1 on the TDM (4 tensor ops in flight).
    if (wave == 0) {
        tdm_load_tile_f16(xh + (size_t)m0 * D_IN,       lds_addr_of(&sX[0][0][0]), M_TOTAL, TBM);
        tdm_load_tile_f16(wh + (size_t)n0 * D_IN,       lds_addr_of(&sW[0][0][0]), D_OUT,   TBN);
        tdm_load_tile_f16(xh + (size_t)m0 * D_IN + TBK, lds_addr_of(&sX[1][0][0]), M_TOTAL, TBM);
        tdm_load_tile_f16(wh + (size_t)n0 * D_IN + TBK, lds_addr_of(&sW[1][0][0]), D_OUT,   TBN);
    }

#pragma unroll 1
    for (int i = 0; i < NIT; ++i) {
        const int buf = i & 1;

        // Tile i (oldest 2 tensor ops) must have landed; tile i+1 may still fly.
        if (wave == 0) {
            if (i + 1 < NIT) __builtin_amdgcn_s_wait_tensorcnt(2);
            else             __builtin_amdgcn_s_wait_tensorcnt(0);
        }
        __syncthreads();

        // B fragments: lane = 16*khalf + n holds W[n][k..k+15] contiguous.
        v16h bfrag[2];
#pragma unroll
        for (int j = 0; j < 2; ++j) {
            const int n = wn * 32 + j * 16 + l16;
            union { v16h v; v8h h[2]; } u;
            u.h[0] = *(const v8h*)&sW[buf][n][lhalf * 16];
            u.h[1] = *(const v8h*)&sW[buf][n][lhalf * 16 + 8];
            bfrag[j] = u.v;
        }
        // A fragments: lanes 0-15 hold K 0..15; lanes 16-31 hold K 16..31.
        v16h afrag[4];
#pragma unroll
        for (int ii = 0; ii < 4; ++ii) {
            const int m = wm * 64 + ii * 16 + l16;
            union { v16h v; v8h h[2]; } u;
            u.h[0] = *(const v8h*)&sX[buf][m][lhalf * 16];
            u.h[1] = *(const v8h*)&sX[buf][m][lhalf * 16 + 8];
            afrag[ii] = u.v;
        }
#pragma unroll
        for (int ii = 0; ii < 4; ++ii)
#pragma unroll
            for (int j = 0; j < 2; ++j)
                acc[ii][j] = __builtin_amdgcn_wmma_f32_16x16x32_f16(
                    false, afrag[ii], false, bfrag[j],
                    (short)0, acc[ii][j], false, false);

        __syncthreads();   // everyone done reading buf before TDM refills it

        if (wave == 0 && i + 2 < NIT) {
            const int k2 = (i + 2) * TBK;
            tdm_load_tile_f16(xh + (size_t)m0 * D_IN + k2, lds_addr_of(&sX[buf][0][0]), M_TOTAL, TBM);
            tdm_load_tile_f16(wh + (size_t)n0 * D_IN + k2, lds_addr_of(&sW[buf][0][0]), D_OUT,   TBN);
        }
    }

    // Epilogue: C/D layout VGPR r <-> row r + 8*(lane>=16); lane&15 <-> column.
#pragma unroll
    for (int ii = 0; ii < 4; ++ii) {
        const int mbase = m0 + wm * 64 + ii * 16 + lhalf * 8;
#pragma unroll
        for (int j = 0; j < 2; ++j) {
            const int n = n0 + wn * 32 + j * 16 + l16;
#pragma unroll
            for (int r = 0; r < 8; ++r)
                out[(size_t)(mbase + r) * D_OUT + n] = acc[ii][j][r] * sc[j] + bi[j];
        }
    }
}

// ---------------------------------------------------------------------------
// Fallback: fused single-pass kernel (round-1 design) if workspace too small.
// ---------------------------------------------------------------------------
__global__ __launch_bounds__(256)
void qlinear_wmma_fused(const float* __restrict__ x,
                        const int*   __restrict__ w,
                        const float* __restrict__ scale,
                        const float* __restrict__ bias,
                        float*       __restrict__ out)
{
    __shared__ _Float16 sX[FBM][FLDSK];
    __shared__ _Float16 sW[FBN][FLDSK];

    const int tid   = threadIdx.x;
    const int lane  = tid & 31;
    const int wave  = tid >> 5;
    const int wm    = wave >> 2;
    const int wn    = wave & 3;
    const int lhalf = lane >> 4;
    const int l16   = lane & 15;

    const int m0 = blockIdx.y * FBM;
    const int n0 = blockIdx.x * FBN;

    float sc[2], bi[2];
#pragma unroll
    for (int j = 0; j < 2; ++j) {
        const int n = n0 + wn * 32 + j * 16 + l16;
        sc[j] = scale[n];
        bi[j] = bias[n];
    }

    v8f acc[4][2] = {};

#pragma unroll 1
    for (int k0 = 0; k0 < D_IN; k0 += FBK) {
#pragma unroll
        for (int i = 0; i < 8; ++i) {
            const int f  = tid + i * 256;
            const int r  = f >> 4;
            const int c4 = f & 15;
            const v4f xv = *(const v4f*)(x + (size_t)(m0 + r) * D_IN + k0 + c4 * 4);
            _Float16* dst = &sX[r][c4 * 4];
            dst[0] = (_Float16)xv.x; dst[1] = (_Float16)xv.y;
            dst[2] = (_Float16)xv.z; dst[3] = (_Float16)xv.w;
        }
#pragma unroll
        for (int i = 0; i < 8; ++i) {
            const int f  = tid + i * 256;
            const int r  = f >> 4;
            const int c4 = f & 15;
            const v4i wv = *(const v4i*)(w + (size_t)(n0 + r) * D_IN + k0 + c4 * 4);
            _Float16* dst = &sW[r][c4 * 4];
            dst[0] = (_Float16)wv.x; dst[1] = (_Float16)wv.y;
            dst[2] = (_Float16)wv.z; dst[3] = (_Float16)wv.w;
        }
        __syncthreads();

#pragma unroll
        for (int ks = 0; ks < FBK; ks += 32) {
            v16h bfrag[2];
#pragma unroll
            for (int j = 0; j < 2; ++j) {
                const int n  = wn * 32 + j * 16 + l16;
                const int kk = ks + lhalf * 16;
                union { v16h v; v8h h[2]; } u;
                u.h[0] = *(const v8h*)&sW[n][kk];
                u.h[1] = *(const v8h*)&sW[n][kk + 8];
                bfrag[j] = u.v;
            }
            v16h afrag[4];
#pragma unroll
            for (int i = 0; i < 4; ++i) {
                const int m = wm * 64 + i * 16 + l16;
                union { v16h v; v8h h[2]; } u;
                u.h[0] = *(const v8h*)&sX[m][ks + lhalf * 16];
                u.h[1] = *(const v8h*)&sX[m][ks + lhalf * 16 + 8];
                afrag[i] = u.v;
            }
#pragma unroll
            for (int i = 0; i < 4; ++i)
#pragma unroll
                for (int j = 0; j < 2; ++j)
                    acc[i][j] = __builtin_amdgcn_wmma_f32_16x16x32_f16(
                        false, afrag[i], false, bfrag[j],
                        (short)0, acc[i][j], false, false);
        }
        __syncthreads();
    }

#pragma unroll
    for (int i = 0; i < 4; ++i) {
        const int mbase = m0 + wm * 64 + i * 16 + lhalf * 8;
#pragma unroll
        for (int j = 0; j < 2; ++j) {
            const int n = n0 + wn * 32 + j * 16 + l16;
#pragma unroll
            for (int r = 0; r < 8; ++r)
                out[(size_t)(mbase + r) * D_OUT + n] = acc[i][j][r] * sc[j] + bi[j];
        }
    }
}

// ---------------------------------------------------------------------------
extern "C" void kernel_launch(void* const* d_in, const int* in_sizes, int n_in,
                              void* d_out, int out_size, void* d_ws, size_t ws_size,
                              hipStream_t stream) {
    const float* x     = (const float*)d_in[0];
    const int*   w     = (const int*)d_in[1];
    const float* scale = (const float*)d_in[2];
    const float* bias  = (const float*)d_in[3];
    float*       out   = (float*)d_out;

    const size_t XH_ELEMS = (size_t)M_TOTAL * D_IN;   // 33,554,432
    const size_t WH_ELEMS = (size_t)D_OUT  * D_IN;    // 16,777,216
    const size_t WS_NEEDED = (XH_ELEMS + WH_ELEMS) * sizeof(_Float16); // ~96 MB

    dim3 grid(D_OUT / TBN, M_TOTAL / TBM);   // (32, 64)

    if (ws_size >= WS_NEEDED) {
        _Float16* xh = (_Float16*)d_ws;
        _Float16* wh = xh + XH_ELEMS;

        const int xn4 = (int)(XH_ELEMS / 4);
        const int wn4 = (int)(WH_ELEMS / 4);
        cvt_f32_to_f16<<<(xn4 + 255) / 256, 256, 0, stream>>>(x, xh, xn4);
        cvt_i32_to_f16<<<(wn4 + 255) / 256, 256, 0, stream>>>(w, wh, wn4);

        qlinear_wmma_tdm<<<grid, 256, 0, stream>>>(xh, wh, scale, bias, out);
    } else {
        qlinear_wmma_fused<<<grid, 256, 0, stream>>>(x, w, scale, bias, out);
    }
}
